// LMModel_transformer_4406636446324
// MI455X (gfx1250) — matrix operand order repaired and checked by reading, not compiled, not run on hardware
//
#include <hip/hip_runtime.h>
#include <hip/hip_bf16.h>
#include <cmath>

typedef _Float16 v16h __attribute__((ext_vector_type(16)));
typedef _Float16 v8h  __attribute__((ext_vector_type(8)));
typedef _Float16 v4h  __attribute__((ext_vector_type(4)));
typedef float    v8f  __attribute__((ext_vector_type(8)));

constexpr int NVOC_ = 32000, D_ = 512, H_ = 8, NL_ = 4, L_ = 2048, Bb_ = 4;
constexpr int DFF_ = 2048, DK_ = 64, T_ = L_ * Bb_;   // 8192 tokens

// ---------------- 64x64 per-wave GEMM tile: C = A(f16,[M,K]) * B(f16,[N,K])^T ----------------
// A frag (16x32): lane m = lane&15, 16 contiguous K at (lane>>4)*16.
// B frag (32x16): lane n = lane&15, 16 contiguous K at (lane>>4)*16.
// C frag: element e -> M = e + 8*(lane>>4), N = lane&15.
template <bool RELU, bool CAUSAL_SKIP>
__device__ __forceinline__ void gemm_tile(const _Float16* __restrict__ A, long lda,
                                          const _Float16* __restrict__ Bm, long ldb,
                                          float* __restrict__ C32, _Float16* __restrict__ C16,
                                          long ldc, const float* __restrict__ bias,
                                          int K, int m0, int n0) {
  if (CAUSAL_SKIP && n0 > m0 + 63) return;        // tile fully above diagonal -> never read
  const int lane = threadIdx.x & 31;
  const int nl = lane & 15;
  const int kh = lane >> 4;

  // hoist row base pointers; inner loop only bumps them by 32 elements
  const _Float16* pa[4];
  const _Float16* pb[4];
#pragma unroll
  for (int i = 0; i < 4; ++i) pa[i] = A + (long)(m0 + i * 16 + nl) * lda + kh * 16;
#pragma unroll
  for (int j = 0; j < 4; ++j) pb[j] = Bm + (long)(n0 + j * 16 + nl) * ldb + kh * 16;

  v8f acc[4][4] = {};

  for (int k0 = 0; k0 < K; k0 += 32) {
    v16h af[4], bf[4];
#pragma unroll
    for (int i = 0; i < 4; ++i) {
      v8h lo = *(const v8h*)pa[i];
      v8h hi = *(const v8h*)(pa[i] + 8);
      af[i] = __builtin_shufflevector(lo, hi, 0, 1, 2, 3, 4, 5, 6, 7,
                                      8, 9, 10, 11, 12, 13, 14, 15);
      pa[i] += 32;
    }
#pragma unroll
    for (int j = 0; j < 4; ++j) {
      bf[j] = *(const v16h*)pb[j];
      pb[j] += 32;
    }
#pragma unroll
    for (int i = 0; i < 4; ++i)
#pragma unroll
      for (int j = 0; j < 4; ++j)
        acc[i][j] = __builtin_amdgcn_wmma_f32_16x16x32_f16(
            false, af[i], false, bf[j], (short)0, acc[i][j], false, false);
  }

#pragma unroll
  for (int j = 0; j < 4; ++j) {
    const int n = n0 + j * 16 + nl;
    const float bv = bias ? bias[n] : 0.f;
#pragma unroll
    for (int i = 0; i < 4; ++i) {
#pragma unroll
      for (int e = 0; e < 8; ++e) {
        const int m = m0 + i * 16 + kh * 8 + e;
        float v = acc[i][j][e] + bv;
        if (RELU) v = fmaxf(v, 0.f);
        if (C32) C32[(size_t)m * ldc + n] = v;
        if (C16) C16[(size_t)m * ldc + n] = (_Float16)v;
      }
    }
  }
}

// ---------------- generic GEMM kernel: 4 waves/block, one 64x64 tile per wave ----------------
template <bool RELU>
__global__ void __launch_bounds__(128) k_gemm(const _Float16* A, long lda,
                                              const _Float16* Bm, long ldb,
                                              float* C32, _Float16* C16, long ldc,
                                              const float* bias, int K, int tilesN, int total) {
  const int tile = blockIdx.x * 4 + (threadIdx.x >> 5);
  if (tile >= total) return;
  const int tm = tile / tilesN, tn = tile % tilesN;
  gemm_tile<RELU, false>(A, lda, Bm, ldb, C32, C16, ldc, bias, K, tm * 64, tn * 64);
}

// ---------------- attention: scores = Q K^T (per head/batch pair), causal tiles skipped ------
__global__ void __launch_bounds__(128) k_scores(const _Float16* Q, const _Float16* Km,
                                                float* S, int pair0) {
  const int tN = L_ / 64;                           // 32
  const int tile = blockIdx.x * 4 + (threadIdx.x >> 5);
  if (tile >= tN * tN) return;
  const int q = pair0 + blockIdx.y;
  const int h = q >> 2, b = q & 3;                  // q = h*B + b
  const _Float16* A = Q + b * D_ + h * DK_;
  const _Float16* Bm = Km + b * D_ + h * DK_;
  float* C = S + (size_t)blockIdx.y * L_ * L_;
  const int tm = tile / tN, tn = tile % tN;
  gemm_tile<false, true>(A, (long)Bb_ * D_, Bm, (long)Bb_ * D_,
                         C, nullptr, L_, nullptr, DK_, tm * 64, tn * 64);
}

// ---------------- causal softmax (scale 1/8 folded into exp), fp32 in / f16 probs out --------
__global__ void __launch_bounds__(256) k_softmax(const float* S, _Float16* Pr) {
  const int l = blockIdx.x, tid = threadIdx.x;
  const float* row = S + (size_t)blockIdx.y * L_ * L_ + (size_t)l * L_;
  _Float16* prow = Pr + (size_t)blockIdx.y * L_ * L_ + (size_t)l * L_;
  const int valid = l + 1;
  __shared__ float red[256];
  float m = -3.4e38f;
  for (int j = tid; j < valid; j += 256) m = fmaxf(m, row[j]);
  red[tid] = m; __syncthreads();
  for (int s = 128; s > 0; s >>= 1) { if (tid < s) red[tid] = fmaxf(red[tid], red[tid + s]); __syncthreads(); }
  m = red[0]; __syncthreads();
  float sum = 0.f;
  for (int j = tid; j < valid; j += 256) sum += expf((row[j] - m) * 0.125f);
  red[tid] = sum; __syncthreads();
  for (int s = 128; s > 0; s >>= 1) { if (tid < s) red[tid] += red[tid + s]; __syncthreads(); }
  const float inv = 1.f / red[0];
  for (int j = tid; j < L_; j += 256)
    prow[j] = (j < valid) ? (_Float16)(expf((row[j] - m) * 0.125f) * inv) : (_Float16)0.f;
}

// ---------------- ctx = P @ V  (V pre-transposed to [b][d][l]) -------------------------------
__global__ void __launch_bounds__(128) k_ctx(const _Float16* Pm, const _Float16* Vt,
                                             _Float16* Ctx, int pair0) {
  const int tile = blockIdx.x * 4 + (threadIdx.x >> 5);
  if (tile >= L_ / 64) return;
  const int q = pair0 + blockIdx.y;
  const int h = q >> 2, b = q & 3;
  const _Float16* A = Pm + (size_t)blockIdx.y * L_ * L_;
  const _Float16* Bm = Vt + ((long)b * D_ + h * DK_) * L_;
  _Float16* C = Ctx + b * D_ + h * DK_;                      // token layout [l][b][D]
  gemm_tile<false, false>(A, L_, Bm, L_, nullptr, C, (long)Bb_ * D_,
                          nullptr, L_, tile * 64, 0);
}

// ---------------- f32 -> f16 weight conversion (4 elems/thread) ------------------------------
__global__ void __launch_bounds__(256) k_cvt(const float* __restrict__ s,
                                             _Float16* __restrict__ d, int n4) {
  const int i = blockIdx.x * 256 + threadIdx.x;
  if (i >= n4) return;
  const float4 v = ((const float4*)s)[i];
  v4h o;
  o[0] = (_Float16)v.x; o[1] = (_Float16)v.y; o[2] = (_Float16)v.z; o[3] = (_Float16)v.w;
  ((v4h*)d)[i] = o;
}

// ---------------- V[l,b,n] (f16) -> Vt[b][n][l] (f16) ----------------------------------------
__global__ void __launch_bounds__(256) k_vtrans(const _Float16* V, _Float16* Vt) {
  const size_t idx = (size_t)blockIdx.x * 256 + threadIdx.x;   // total T_*D_
  const int l = (int)(idx & (L_ - 1));
  const int n = (int)((idx >> 11) & (D_ - 1));
  const int b = (int)(idx >> 20);
  Vt[idx] = V[((size_t)l * Bb_ + b) * D_ + n];
}

// ---------------- embedding * sqrt(D) + sinusoidal PE ----------------------------------------
__global__ void __launch_bounds__(256) k_embed(const int* tok, const float* emb,
                                               float* x32, _Float16* x16) {
  const int t = blockIdx.x, tid = threadIdx.x;
  const int id = tok[t];
  const int l = t >> 2;                        // t = l*B + b
  const size_t base = (size_t)t * D_;
  for (int d = tid; d < D_; d += 256) {
    float v = emb[(size_t)id * D_ + d] * 22.627416997969522f;   // sqrt(512)
    const int i = d >> 1;
    const float div = expf(-(float)(2 * i) * (9.210340371976184f / (float)D_));
    const float ang = (float)l * div;
    const float pe = (d & 1) ? cosf(ang) : sinf(ang);
    const float o = v + pe;
    x32[base + d] = o;
    x16[base + d] = (_Float16)o;
  }
}

// ---------------- residual add + torch-style LayerNorm (unbiased std, /(std+eps)) ------------
__global__ void __launch_bounds__(256) k_ln(const float* X, const float* Y,
                                            const float* ga, const float* gb,
                                            float* o32, _Float16* o16) {
  const int t = blockIdx.x, tid = threadIdx.x;
  __shared__ float red[256];
  const size_t base = (size_t)t * D_;
  float v0 = X[base + tid], v1 = X[base + tid + 256];
  if (Y) { v0 += Y[base + tid]; v1 += Y[base + tid + 256]; }
  red[tid] = v0 + v1; __syncthreads();
  for (int s = 128; s > 0; s >>= 1) { if (tid < s) red[tid] += red[tid + s]; __syncthreads(); }
  const float mean = red[0] * (1.f / D_);
  __syncthreads();
  const float d0 = v0 - mean, d1 = v1 - mean;
  red[tid] = d0 * d0 + d1 * d1; __syncthreads();
  for (int s = 128; s > 0; s >>= 1) { if (tid < s) red[tid] += red[tid + s]; __syncthreads(); }
  const float var = red[0] * (1.f / (D_ - 1));
  const float inv = 1.f / (sqrtf(var) + 1e-6f);
  const float r0 = ga[tid] * d0 * inv + gb[tid];
  const float r1 = ga[tid + 256] * d1 * inv + gb[tid + 256];
  if (o32) { o32[base + tid] = r0; o32[base + tid + 256] = r1; }
  if (o16) { o16[base + tid] = (_Float16)r0; o16[base + tid + 256] = (_Float16)r1; }
}

// =============================================================================================
extern "C" void kernel_launch(void* const* d_in, const int* in_sizes, int n_in,
                              void* d_out, int out_size, void* d_ws, size_t ws_size,
                              hipStream_t stream) {
  (void)in_sizes; (void)n_in; (void)out_size;
  const int*   tok  = (const int*)d_in[0];
  const float* emb  = (const float*)d_in[1];
  const float* Wq   = (const float*)d_in[2];
  const float* Wk   = (const float*)d_in[3];
  const float* Wv   = (const float*)d_in[4];
  const float* Wo   = (const float*)d_in[5];
  const float* w1   = (const float*)d_in[6];
  const float* b1   = (const float*)d_in[7];
  const float* w2   = (const float*)d_in[8];
  const float* b2   = (const float*)d_in[9];
  const float* ln1a = (const float*)d_in[10];
  const float* ln1b = (const float*)d_in[11];
  const float* ln2a = (const float*)d_in[12];
  const float* ln2b = (const float*)d_in[13];
  const float* fna  = (const float*)d_in[14];
  const float* fnb  = (const float*)d_in[15];
  const float* decW = (const float*)d_in[16];
  const float* decb = (const float*)d_in[17];
  float* out = (float*)d_out;

  // ---- workspace carve-up ----
  char* ws = (char*)d_ws;
  size_t off = 0;
  auto take = [&](size_t bytes) { char* p = ws + off; off += (bytes + 255) & ~(size_t)255; return p; };
  float*    x32   = (float*)take((size_t)T_ * D_ * 4);
  _Float16* x16   = (_Float16*)take((size_t)T_ * D_ * 2);
  float*    t32   = (float*)take((size_t)T_ * D_ * 4);
  _Float16* q16   = (_Float16*)take((size_t)T_ * D_ * 2);
  _Float16* k16   = (_Float16*)take((size_t)T_ * D_ * 2);
  _Float16* v16b  = (_Float16*)take((size_t)T_ * D_ * 2);
  _Float16* vt16  = (_Float16*)take((size_t)T_ * D_ * 2);
  _Float16* ctx16 = (_Float16*)take((size_t)T_ * D_ * 2);
  _Float16* ff16  = q16;   // FFN hidden [T_,DFF_] f16 = 33.5MB, aliases dead q16..vt16 (33.5MB)
  // f16 weight copies (converted once per launch)
  _Float16* wq16 = (_Float16*)take((size_t)NL_ * D_ * D_ * 2);
  _Float16* wk16 = (_Float16*)take((size_t)NL_ * D_ * D_ * 2);
  _Float16* wv16 = (_Float16*)take((size_t)NL_ * D_ * D_ * 2);
  _Float16* wo16 = (_Float16*)take((size_t)NL_ * D_ * D_ * 2);
  _Float16* w116 = (_Float16*)take((size_t)NL_ * DFF_ * D_ * 2);
  _Float16* w216 = (_Float16*)take((size_t)NL_ * D_ * DFF_ * 2);
  _Float16* dW16 = (_Float16*)take((size_t)NVOC_ * D_ * 2);
  const size_t base = off;
  const size_t pairBytes = (size_t)L_ * L_ * 4 + (size_t)L_ * L_ * 2;   // scores f32 + probs f16
  int P = 1;
  { const int cands[6] = {32, 16, 8, 4, 2, 1};
    for (int ci = 0; ci < 6; ++ci)
      if (base + (size_t)cands[ci] * pairBytes <= ws_size) { P = cands[ci]; break; } }
  float*    sc32 = (float*)(ws + base);
  _Float16* at16 = (_Float16*)(ws + base + (size_t)P * L_ * L_ * 4);

  // ---- one-shot weight conversion to f16 ----
  auto cvt = [&](const float* s, _Float16* d, size_t n) {
    const int n4 = (int)(n / 4);
    k_cvt<<<dim3((n4 + 255) / 256), dim3(256), 0, stream>>>(s, d, n4);
  };
  cvt(Wq, wq16, (size_t)NL_ * D_ * D_);
  cvt(Wk, wk16, (size_t)NL_ * D_ * D_);
  cvt(Wv, wv16, (size_t)NL_ * D_ * D_);
  cvt(Wo, wo16, (size_t)NL_ * D_ * D_);
  cvt(w1, w116, (size_t)NL_ * DFF_ * D_);
  cvt(w2, w216, (size_t)NL_ * D_ * DFF_);
  cvt(decW, dW16, (size_t)NVOC_ * D_);

  // ---- embedding + positional encoding ----
  k_embed<<<dim3(T_), dim3(256), 0, stream>>>(tok, emb, x32, x16);

  const int tNd = D_ / 64;                          // 8
  const int totD = (T_ / 64) * tNd;                 // 1024 tiles -> 256 blocks

  for (int i = 0; i < NL_; ++i) {
    const _Float16* wq = wq16 + (size_t)i * D_ * D_;
    const _Float16* wk = wk16 + (size_t)i * D_ * D_;
    const _Float16* wv = wv16 + (size_t)i * D_ * D_;
    const _Float16* wo = wo16 + (size_t)i * D_ * D_;
    const _Float16* W1 = w116 + (size_t)i * DFF_ * D_;
    const _Float16* W2 = w216 + (size_t)i * D_ * DFF_;
    const float* B1 = b1 + (size_t)i * DFF_;
    const float* B2 = b2 + (size_t)i * D_;

    // Q, K, V projections (f16 outputs)
    k_gemm<false><<<dim3(totD / 4), dim3(128), 0, stream>>>(
        x16, D_, wq, D_, nullptr, q16, D_, nullptr, D_, tNd, totD);
    k_gemm<false><<<dim3(totD / 4), dim3(128), 0, stream>>>(
        x16, D_, wk, D_, nullptr, k16, D_, nullptr, D_, tNd, totD);
    k_gemm<false><<<dim3(totD / 4), dim3(128), 0, stream>>>(
        x16, D_, wv, D_, nullptr, v16b, D_, nullptr, D_, tNd, totD);
    k_vtrans<<<dim3((T_ * D_) / 256), dim3(256), 0, stream>>>(v16b, vt16);

    // attention, P head-batch pairs at a time
    for (int g = 0; g < 32 / P; ++g) {
      const int pair0 = g * P;
      k_scores<<<dim3((L_ / 64) * (L_ / 64) / 4, P), dim3(128), 0, stream>>>(q16, k16, sc32, pair0);
      k_softmax<<<dim3(L_, P), dim3(256), 0, stream>>>(sc32, at16);
      k_ctx<<<dim3((L_ / 64) / 4, P), dim3(128), 0, stream>>>(at16, vt16, ctx16, pair0);
    }

    // output projection + residual LN1
    k_gemm<false><<<dim3(totD / 4), dim3(128), 0, stream>>>(
        ctx16, D_, wo, D_, t32, nullptr, D_, nullptr, D_, tNd, totD);
    k_ln<<<dim3(T_), dim3(256), 0, stream>>>(x32, t32, ln1a + i * D_, ln1b + i * D_, x32, x16);

    // FFN: relu(x W1^T + b1) W2^T + b2, then residual LN2
    const int tNf = DFF_ / 64;                      // 32
    const int totF = (T_ / 64) * tNf;               // 4096 tiles
    k_gemm<true><<<dim3(totF / 4), dim3(128), 0, stream>>>(
        x16, D_, W1, D_, nullptr, ff16, DFF_, B1, D_, tNf, totF);
    k_gemm<false><<<dim3(totD / 4), dim3(128), 0, stream>>>(
        ff16, DFF_, W2, DFF_, t32, nullptr, D_, B2, DFF_, tNd, totD);
    k_ln<<<dim3(T_), dim3(256), 0, stream>>>(x32, t32, ln2a + i * D_, ln2b + i * D_, x32, x16);
  }

  // final LN + decoder
  k_ln<<<dim3(T_), dim3(256), 0, stream>>>(x32, nullptr, fna, fnb, nullptr, x16);
  { const int tNv = NVOC_ / 64;                     // 500
    const int totV = (T_ / 64) * tNv;               // 64000 tiles
    k_gemm<false><<<dim3(totV / 4), dim3(128), 0, stream>>>(
        x16, D_, dW16, D_, out, nullptr, NVOC_, decb, D_, tNv, totV); }
}
